// Hierarchical_Model_73856257622064
// MI455X (gfx1250) — matrix operand order repaired and checked by reading, not compiled, not run on hardware
//
#include <hip/hip_runtime.h>
#include <hip/hip_bf16.h>
#include <stdint.h>

// ---------------- problem constants ----------------
#define V_   100000
#define E_   300
#define H_   512
#define T_   5
#define S_   4096
#define KP_  320          // E padded to multiple of 32 for WMMA K-loop
#define G4H_ 2048         // 4*H
#define STARTTAG 3
#define STOPTAG  4
#define NEGV (-10000.0f)

#define NWG_ 16           // workgroups per LSTM direction
#define UPW_ 32           // hidden units per workgroup (512/16)
#define RPW_ 128          // gate rows per workgroup (4*UPW_)

typedef __attribute__((ext_vector_type(8)))  __bf16 bf16x8;
typedef __attribute__((ext_vector_type(16))) __bf16 bf16x16;
typedef __attribute__((ext_vector_type(8)))  float  f32x8;

__device__ __forceinline__ __bf16 f2bf(float f) {
    unsigned u = __float_as_uint(f);
    unsigned r = u + 0x7FFFu + ((u >> 16) & 1u);   // round to nearest even
    return __builtin_bit_cast(__bf16, (unsigned short)(r >> 16));
}

// ---------------- init: zero h double-buffers and barrier counters ----------------
__global__ void k_init(float* hbuf, unsigned int* cnt) {
    int i = blockIdx.x * blockDim.x + threadIdx.x;
    if (i < 2 * 2 * H_) hbuf[i] = 0.0f;     // [dir][2 buffers][H]
    if (i < 128)        cnt[i]  = 0u;       // per-direction barrier counters (padded)
}

// ---------------- prep: gather + f32->bf16 + K padding ----------------
__global__ void k_prep_emb(const int* __restrict__ sentence,
                           const float* __restrict__ embedding,
                           __bf16* __restrict__ embB) {
    int idx = blockIdx.x * blockDim.x + threadIdx.x;      // over S_*KP_
    if (idx >= S_ * KP_) return;
    int s = idx / KP_, k = idx - s * KP_;
    float v = (k < E_) ? embedding[(size_t)sentence[s] * E_ + k] : 0.0f;
    embB[idx] = f2bf(v);
}

__global__ void k_prep_w(const float* __restrict__ w_ih_f,
                         const float* __restrict__ w_ih_b,
                         __bf16* __restrict__ wB) {
    int idx = blockIdx.x * blockDim.x + threadIdx.x;      // over 2*G4H_*KP_
    if (idx >= 2 * G4H_ * KP_) return;
    int dir = idx / (G4H_ * KP_);
    int rem = idx - dir * (G4H_ * KP_);
    int j = rem / KP_, k = rem - j * KP_;
    const float* src = dir ? w_ih_b : w_ih_f;
    float v = (k < E_) ? src[(size_t)j * E_ + k] : 0.0f;
    wB[idx] = f2bf(v);
}

__global__ void k_prep_bias(const float* __restrict__ b_ih_f, const float* __restrict__ b_hh_f,
                            const float* __restrict__ b_ih_b, const float* __restrict__ b_hh_b,
                            float* __restrict__ biasB) {
    int idx = blockIdx.x * blockDim.x + threadIdx.x;      // over 2*G4H_
    if (idx >= 2 * G4H_) return;
    int dir = idx / G4H_, j = idx - dir * G4H_;
    biasB[idx] = dir ? (b_ih_b[j] + b_hh_b[j]) : (b_ih_f[j] + b_hh_f[j]);
}

// ---------------- input projection GEMM: xp[dir][s][4H] via WMMA bf16 ----------------
// grid: x = G4H_/16/8 = 16, y = S_/16 = 256, z = 2 (direction); block = 256 (8 waves)
__global__ __launch_bounds__(256) void k_xproj(const __bf16* __restrict__ embB,
                                               const __bf16* __restrict__ wB,
                                               const float* __restrict__ biasB,
                                               float* __restrict__ xp) {
    int lane = threadIdx.x & 31;
    int wv   = threadIdx.x >> 5;
    int dir  = blockIdx.z;
    int n0   = (blockIdx.x * 8 + wv) * 16;   // column tile in 4H
    int m0   = blockIdx.y * 16;              // row tile in S
    int lo16 = lane & 15;
    int hi   = lane >> 4;

    const __bf16* A = embB + (size_t)(m0 + lo16) * KP_;
    const __bf16* B = wB + (size_t)dir * G4H_ * KP_ + (size_t)(n0 + lo16) * KP_;

    f32x8 acc = {};
#pragma unroll
    for (int kt = 0; kt < KP_; kt += 32) {
        // A: 16x32 tile — lane holds row M=lo16; this code loads 16 contiguous K at 16*hi (B matches)
        bf16x16 av = *(const bf16x16*)(A + kt + 16 * hi);
        // B: 32x16 tile — lane holds column N=lo16; same K order as A: 16 contiguous K at 16*hi
        bf16x16 bv = *(const bf16x16*)(B + kt + 16 * hi);
        acc = __builtin_amdgcn_wmma_f32_16x16x32_bf16(false, av, false, bv,
                                                      (short)0, acc, false, false);
    }
    float bias = biasB[dir * G4H_ + n0 + lo16];
    float* out = xp + (size_t)dir * S_ * G4H_;
    int n = n0 + lo16;
#pragma unroll
    for (int r = 0; r < 8; ++r)
        out[(size_t)(m0 + r + 8 * hi) * G4H_ + n] = acc[r] + bias;
}

// ---------------- LSTM recurrence: 16 WGs per direction, W_hh slice LDS-resident ----------------
// grid.x = 32 (dir = blockIdx.x>>4); block = 256
__global__ __launch_bounds__(256) void k_lstm(const float* __restrict__ w_hh_f,
                                              const float* __restrict__ w_hh_b,
                                              const float* __restrict__ xp,
                                              float* __restrict__ hs,
                                              float* hbuf,
                                              unsigned int* barrier_cnt) {
    __shared__ float Wsh[RPW_ * H_];   // 256 KB: this WG's 128 gate rows
    __shared__ float hsh[H_];
    __shared__ float gsh[RPW_];

    int dir = blockIdx.x >> 4;
    int wg  = blockIdx.x & 15;
    int u0  = wg * UPW_;
    int tid = threadIdx.x;
    int lane = tid & 31, wv = tid >> 5;

    const float* Whh = dir ? w_hh_b : w_hh_f;
    const float* xpd = xp + (size_t)dir * S_ * G4H_;
    float* hsd = hs + (size_t)dir * S_ * H_;
    float* hb  = hbuf + dir * 2 * H_;          // double buffer [2][H]
    unsigned int* cnt = barrier_cnt + dir * 64;

    // stage this WG's W_hh rows: r -> global row (r>>5)*512 + u0 + (r&31)
    for (int i = tid; i < RPW_ * H_; i += 256) {
        int r = i / H_, k = i - r * H_;
        Wsh[i] = Whh[(size_t)((r >> 5) * H_ + u0 + (r & 31)) * H_ + k];
    }
    float c = 0.0f;                            // cell state (valid in tid<UPW_)
    __syncthreads();

    for (int t = 0; t < S_; ++t) {
        int tt = dir ? (S_ - 1 - t) : t;
        const volatile float* hr = hb + (t & 1) * H_;        // read buffer (L0-bypass)
        float* hw = hb + ((t + 1) & 1) * H_;                 // write buffer
        for (int k = tid; k < H_; k += 256) hsh[k] = hr[k];
        __syncthreads();

        // 128 dot products of length 512; 2 lanes per row
        int r    = wv * 16 + (lane >> 1);
        int half = lane & 1;
        const float* wrow  = &Wsh[r * H_ + half * 256];
        const float* hpart = &hsh[half * 256];
        float acc = 0.0f;
#pragma unroll 8
        for (int k = 0; k < 256; ++k) acc += wrow[k] * hpart[k];
        acc += __shfl_xor(acc, 1, 32);
        if (half == 0)
            gsh[r] = acc + xpd[(size_t)tt * G4H_ + ((r >> 5) * H_ + u0 + (r & 31))];
        __syncthreads();

        if (tid < UPW_) {
            float gi = gsh[tid], gf = gsh[32 + tid], gg = gsh[64 + tid], go = gsh[96 + tid];
            float si = 1.0f / (1.0f + __expf(-gi));
            float sf = 1.0f / (1.0f + __expf(-gf));
            float so = 1.0f / (1.0f + __expf(-go));
            c = sf * c + si * tanhf(gg);
            float hn = so * tanhf(c);
            hw[u0 + tid] = hn;
            hsd[(size_t)tt * H_ + u0 + tid] = hn;
        }
        __threadfence();
        __syncthreads();
        if (tid == 0) {
            atomicAdd(cnt, 1u);
            unsigned target = (unsigned)(t + 1) * NWG_;
            while (__hip_atomic_load(cnt, __ATOMIC_ACQUIRE, __HIP_MEMORY_SCOPE_AGENT) < target) {}
        }
        __syncthreads();
    }
}

// ---------------- tag projection: feats[s][tag] ----------------
__global__ void k_feats(const float* __restrict__ hs,
                        const float* __restrict__ w_tag,
                        const float* __restrict__ b_tag,
                        float* __restrict__ feats) {
    int idx = blockIdx.x * blockDim.x + threadIdx.x;
    if (idx >= S_ * T_) return;
    int s = idx / T_, tg = idx - s * T_;
    const float* hf = hs + (size_t)s * H_;
    const float* hb = hs + (size_t)(S_ + s) * H_;
    const float* w  = w_tag + (size_t)tg * 2 * H_;
    float acc = b_tag[tg];
    for (int k = 0; k < H_; ++k) acc += hf[k] * w[k];
    for (int k = 0; k < H_; ++k) acc += hb[k] * w[H_ + k];
    feats[idx] = acc;
}

// ---------------- CRF: gold score + logsumexp forward scan ----------------
__global__ __launch_bounds__(256) void k_crf(const float* __restrict__ feats,
                                             const float* __restrict__ trans,
                                             const int* __restrict__ tags,
                                             float* __restrict__ out) {
    __shared__ float red[256];
    __shared__ float goldsh;
    int tid = threadIdx.x;
    float g = 0.0f;
    for (int t = tid; t < S_; t += 256) {
        int cur  = tags[t];
        int prev = (t == 0) ? STARTTAG : tags[t - 1];
        g += trans[cur * T_ + prev] + feats[t * T_ + cur];
    }
    red[tid] = g;
    __syncthreads();
    for (int s = 128; s > 0; s >>= 1) { if (tid < s) red[tid] += red[tid + s]; __syncthreads(); }
    if (tid == 0) goldsh = red[0] + trans[STOPTAG * T_ + tags[S_ - 1]];
    __syncthreads();

    if (tid < 32) {
        float tr0 = 0, tr1 = 0, tr2 = 0, tr3 = 0, tr4 = 0, trS = NEGV;
        if (tid < T_) {
            tr0 = trans[tid * T_ + 0]; tr1 = trans[tid * T_ + 1]; tr2 = trans[tid * T_ + 2];
            tr3 = trans[tid * T_ + 3]; tr4 = trans[tid * T_ + 4];
            trS = trans[STOPTAG * T_ + tid];
        }
        float alpha = (tid == STARTTAG) ? 0.0f : NEGV;
        for (int t = 0; t < S_; ++t) {
            float a0 = __shfl(alpha, 0, 32), a1 = __shfl(alpha, 1, 32);
            float a2 = __shfl(alpha, 2, 32), a3 = __shfl(alpha, 3, 32);
            float a4 = __shfl(alpha, 4, 32);
            float v0 = a0 + tr0, v1 = a1 + tr1, v2 = a2 + tr2, v3 = a3 + tr3, v4 = a4 + tr4;
            float m = fmaxf(fmaxf(fmaxf(v0, v1), fmaxf(v2, v3)), v4);
            float lse = m + __logf(__expf(v0 - m) + __expf(v1 - m) + __expf(v2 - m) +
                                   __expf(v3 - m) + __expf(v4 - m));
            float fv = (tid < T_) ? feats[t * T_ + tid] : 0.0f;
            alpha = (tid < T_) ? (lse + fv) : NEGV;
        }
        float fin = alpha + trS;
        float b0 = __shfl(fin, 0, 32), b1 = __shfl(fin, 1, 32), b2 = __shfl(fin, 2, 32);
        float b3 = __shfl(fin, 3, 32), b4 = __shfl(fin, 4, 32);
        if (tid == 0) {
            float m = fmaxf(fmaxf(fmaxf(b0, b1), fmaxf(b2, b3)), b4);
            float fs = m + __logf(__expf(b0 - m) + __expf(b1 - m) + __expf(b2 - m) +
                                  __expf(b3 - m) + __expf(b4 - m));
            out[0] = fs - goldsh;
        }
    }
}

// ---------------- launcher ----------------
extern "C" void kernel_launch(void* const* d_in, const int* in_sizes, int n_in,
                              void* d_out, int out_size, void* d_ws, size_t ws_size,
                              hipStream_t stream) {
    const int*   sentence = (const int*)d_in[0];
    const int*   tags     = (const int*)d_in[1];
    const float* embedding= (const float*)d_in[2];
    const float* w_ih_f   = (const float*)d_in[3];
    const float* w_hh_f   = (const float*)d_in[4];
    const float* b_ih_f   = (const float*)d_in[5];
    const float* b_hh_f   = (const float*)d_in[6];
    const float* w_ih_b   = (const float*)d_in[7];
    const float* w_hh_b   = (const float*)d_in[8];
    const float* b_ih_b   = (const float*)d_in[9];
    const float* b_hh_b   = (const float*)d_in[10];
    const float* w_tag    = (const float*)d_in[11];
    const float* b_tag    = (const float*)d_in[12];
    const float* trans    = (const float*)d_in[13];

    char* ws = (char*)d_ws;
    size_t off = 0;
    auto take = [&](size_t bytes) -> char* {
        char* p = ws + off;
        off += (bytes + 255) & ~(size_t)255;
        return p;
    };
    __bf16* embB  = (__bf16*)take((size_t)S_ * KP_ * 2);
    __bf16* wB    = (__bf16*)take((size_t)2 * G4H_ * KP_ * 2);
    float*  biasB = (float*) take((size_t)2 * G4H_ * 4);
    float*  xp    = (float*) take((size_t)2 * S_ * G4H_ * 4);
    float*  hs    = (float*) take((size_t)2 * S_ * H_ * 4);
    float*  hbuf  = (float*) take((size_t)2 * 2 * H_ * 4);
    float*  feats = (float*) take((size_t)S_ * T_ * 4);
    unsigned int* cnt = (unsigned int*)take(128 * 4);

    k_init<<<8, 256, 0, stream>>>(hbuf, cnt);
    k_prep_emb<<<(S_ * KP_ + 255) / 256, 256, 0, stream>>>(sentence, embedding, embB);
    k_prep_w<<<(2 * G4H_ * KP_ + 255) / 256, 256, 0, stream>>>(w_ih_f, w_ih_b, wB);
    k_prep_bias<<<(2 * G4H_ + 255) / 256, 256, 0, stream>>>(b_ih_f, b_hh_f, b_ih_b, b_hh_b, biasB);
    k_xproj<<<dim3(G4H_ / 16 / 8, S_ / 16, 2), 256, 0, stream>>>(embB, wB, biasB, xp);
    k_lstm<<<2 * NWG_, 256, 0, stream>>>(w_hh_f, w_hh_b, xp, hs, hbuf, cnt);
    k_feats<<<(S_ * T_ + 255) / 256, 256, 0, stream>>>(hs, w_tag, b_tag, feats);
    k_crf<<<1, 256, 0, stream>>>(feats, trans, tags, d_out ? (float*)d_out : nullptr);
}
